// RingAttentionExpertModule_51531017617703
// MI455X (gfx1250) — compile-verified, matrix operand order checked by reading, not run on hardware
//
#include <hip/hip_runtime.h>
#include <hip/hip_bf16.h>

typedef unsigned short u16;
typedef unsigned int u32;
typedef __attribute__((ext_vector_type(16))) __bf16 v16bf;
typedef __attribute__((ext_vector_type(8)))  float  v8f;

#define EMBED 3072
#define SEQ   8192
#define FFN_D 12288
#define QKV_D 9216

// Padded LDS row strides (in bf16 elems) -> conflict-free ds_load_b128 fragment reads.
// 40 elems = 80B = 20 banks (gcd(20,64)=4 but 20L mod 64 distinct for L=0..15); rows stay 16B-aligned.
#define PAD32 40
// 136 elems = 272B = 68 banks -> 4L mod 64, distinct for L=0..15.
#define PAD128 136

struct B32x8 { uint4 a, b; };

__device__ inline u16 f2bf(float f) {
  u32 u = __builtin_bit_cast(u32, f);
  u += 0x7fffu + ((u >> 16) & 1u);
  return (u16)(u >> 16);
}

// Async global -> LDS copy of 16 bytes (CDNA5 GLOBAL_LOAD_ASYNC_TO_LDS_B128, GV mode).
// ldsOff = byte offset within the workgroup LDS allocation (low 32 bits of the flat address).
__device__ inline void async_copy_b128(u32 ldsOff, const void* gaddr) {
  asm volatile("global_load_async_to_lds_b128 %0, %1, off"
               :: "v"(ldsOff), "v"(gaddr)
               : "memory");
}
__device__ inline void async_wait0() {
  asm volatile("s_wait_asynccnt 0x0" ::: "memory");
}
__device__ inline u32 lds_off(const void* p) { return (u32)(uintptr_t)p; }

// Load a 16x32 bf16 WMMA A/B fragment from a row-major tile.
// A-frag: lane<16 -> m=lane, K in {0..7,16..23}; lane>=16 -> m=lane-16, K in {8..15,24..31}.
// The symmetric B layout (lane indexes N) uses the identical access pattern.
__device__ inline v16bf load_frag(const u16* base, int rowStride, int row0, int kOff, int lane) {
  const int half = lane >> 4;
  const u16* p = base + (size_t)(row0 + (lane & 15)) * rowStride + kOff + half * 8;
  B32x8 s;
  s.a = *(const uint4*)(p);
  s.b = *(const uint4*)(p + 16);
  return __builtin_bit_cast(v16bf, s);
}

// ---------------------------------------------------------------- small kernels

__global__ __launch_bounds__(256) void cvt_f32_bf16(const float* __restrict__ in,
                                                    u16* __restrict__ out, size_t n) {
  size_t i = ((size_t)blockIdx.x * 256 + threadIdx.x) * 4;
  if (i >= n) return;
  float4 f = *(const float4*)(in + i);
  out[i + 0] = f2bf(f.x); out[i + 1] = f2bf(f.y);
  out[i + 2] = f2bf(f.z); out[i + 3] = f2bf(f.w);
}

__global__ __launch_bounds__(256) void add_embed_kernel(const float* __restrict__ x,
                                                        const float* __restrict__ ee,
                                                        float* __restrict__ xr,
                                                        u16* __restrict__ xb) {
  size_t i = ((size_t)blockIdx.x * 256 + threadIdx.x) * 4;
  int col = (int)(i % EMBED);
  float4 f = *(const float4*)(x + i);
  float4 e = *(const float4*)(ee + col);
  float4 r; r.x = f.x + e.x; r.y = f.y + e.y; r.z = f.z + e.z; r.w = f.w + e.w;
  *(float4*)(xr + i) = r;
  xb[i + 0] = f2bf(r.x); xb[i + 1] = f2bf(r.y);
  xb[i + 2] = f2bf(r.z); xb[i + 3] = f2bf(r.w);
}

__global__ __launch_bounds__(256) void zero_u16_kernel(u16* __restrict__ p, size_t n) {
  size_t i = ((size_t)blockIdx.x * 256 + threadIdx.x) * 8;
  if (i >= n) return;
  uint4 z; z.x = 0; z.y = 0; z.z = 0; z.w = 0;
  *(uint4*)(p + i) = z;
}

// ---------------------------------------------------------------- WMMA GEMM
// C[M,N] = A[M,K](bf16,row-major) * B[N,K](bf16,row-major)^T + bias
// EPI: 0 = bf16 store, 1 = f32 store, 2 = exact-GELU then bf16 store
template <int EPI>
__global__ __launch_bounds__(256)
void gemm_bf16(const u16* __restrict__ A, const u16* __restrict__ B,
               const float* __restrict__ bias, void* __restrict__ Cout,
               int M, int N, int K) {
  const int tid  = threadIdx.x;
  const int lane = tid & 31, wave = tid >> 5;
  const int half = lane >> 4, l16 = lane & 15;
  const int wm = wave >> 1, wn = wave & 1;            // 4x2 wave grid
  const int rowBase = blockIdx.y * 128;
  const int colBase = blockIdx.x * 128;

  __shared__ __align__(16) u16 As[2][128 * PAD32];
  __shared__ __align__(16) u16 Bs[2][128 * PAD32];

  v8f acc[2][4];
#pragma unroll
  for (int mi = 0; mi < 2; mi++)
#pragma unroll
    for (int ni = 0; ni < 4; ni++) acc[mi][ni] = v8f{};

  // async tile fill: each thread copies 2x16B for A and 2x16B for B
  auto async_tile = [&](int kt, int buf) {
#pragma unroll
    for (int j = 0; j < 2; j++) {
      int q = tid + 256 * j;                   // 512 uint4 per operand tile
      int row = q >> 2, c4 = q & 3;            // 4 uint4 per 32-elem row
      async_copy_b128(lds_off(&As[buf][row * PAD32 + c4 * 8]),
                      A + (size_t)(rowBase + row) * K + kt * 32 + c4 * 8);
      async_copy_b128(lds_off(&Bs[buf][row * PAD32 + c4 * 8]),
                      B + (size_t)(colBase + row) * K + kt * 32 + c4 * 8);
    }
  };

  async_tile(0, 0);
  async_wait0();
  __syncthreads();

  const int KT = K >> 5;
  for (int kt = 0; kt < KT; kt++) {
    const int cur = kt & 1;
    if (kt + 1 < KT) async_tile(kt + 1, cur ^ 1);
    v16bf af[2], bfr[4];
#pragma unroll
    for (int mi = 0; mi < 2; mi++) af[mi] = load_frag(As[cur], PAD32, wm * 32 + mi * 16, 0, lane);
#pragma unroll
    for (int ni = 0; ni < 4; ni++) bfr[ni] = load_frag(Bs[cur], PAD32, wn * 64 + ni * 16, 0, lane);
#pragma unroll
    for (int mi = 0; mi < 2; mi++)
#pragma unroll
      for (int ni = 0; ni < 4; ni++)
        acc[mi][ni] = __builtin_amdgcn_wmma_f32_16x16x32_bf16(
            false, af[mi], false, bfr[ni], (short)0, acc[mi][ni], false, false);
    if (kt + 1 < KT) {
      async_wait0();
      __syncthreads();
    }
  }

#pragma unroll
  for (int mi = 0; mi < 2; mi++) {
#pragma unroll
    for (int ni = 0; ni < 4; ni++) {
      const int col = colBase + wn * 64 + ni * 16 + l16;
      const float bv = bias[col];
#pragma unroll
      for (int i = 0; i < 8; i++) {
        const int row = rowBase + wm * 32 + mi * 16 + i + 8 * half;
        float v = acc[mi][ni][i] + bv;
        if (EPI == 2) v = 0.5f * v * (1.0f + erff(v * 0.70710678118654752f));
        if (EPI == 1)
          ((float*)Cout)[(size_t)row * N + col] = v;
        else
          ((u16*)Cout)[(size_t)row * N + col] = f2bf(v);
      }
    }
  }
}

// ---------------------------------------------------------------- WMMA flash attention
// One block: 128 dilated query rows of one (segment, head). Selected length is 1024 for all groups.
__global__ __launch_bounds__(256)
void attn_wmma(const u16* __restrict__ qkv, u16* __restrict__ out,
               int seg, int dil, int off, int headBase) {
  const int tid = threadIdx.x;
  const int lane = tid & 31, wave = tid >> 5;
  const int half = lane >> 4, l16 = lane & 15;
  const int head = headBase + blockIdx.y;
  const size_t ROW = QKV_D;
  const size_t segBase = (size_t)blockIdx.z * seg + off;
  const int qrow0 = blockIdx.x * 128 + wave * 16;

  __shared__ __align__(16) u16 Kt[32 * PAD128];      // [key][dim], padded
  __shared__ __align__(16) u16 Vt[128 * PAD32];      // [dim][key] (transposed), padded
  __shared__ __align__(16) u16 Pb[8 * 16 * PAD32];   // per-wave P staging, padded

  // Q fragments held in registers for the whole kernel
  v16bf qf[4];
  {
    const size_t qpos = segBase + (size_t)(qrow0 + l16) * dil;
    const u16* qp = qkv + qpos * ROW + (size_t)head * 128;
#pragma unroll
    for (int kk = 0; kk < 4; kk++) {
      B32x8 s;
      s.a = *(const uint4*)(qp + kk * 32 + half * 8);
      s.b = *(const uint4*)(qp + kk * 32 + half * 8 + 16);
      qf[kk] = __builtin_bit_cast(v16bf, s);
    }
  }

  v8f o[8];
#pragma unroll
  for (int n = 0; n < 8; n++) o[n] = v8f{};
  float rm[8], rl[8];
#pragma unroll
  for (int i = 0; i < 8; i++) { rm[i] = -3.0e38f; rl[i] = 0.f; }
  const float scale = 0.08838834764831845f;  // 1/sqrt(128)

  for (int kb = 0; kb < 1024; kb += 32) {
    __syncthreads();
    // K tile: straight copy -> async global->LDS (ASYNCcnt-tracked, no VGPR staging)
#pragma unroll
    for (int j = 0; j < 2; j++) {
      int q = tid + 256 * j;
      int key = q >> 4, c4 = q & 15;
      size_t kpos = segBase + (size_t)(kb + key) * dil;
      async_copy_b128(lds_off(&Kt[key * PAD128 + c4 * 8]),
                      qkv + kpos * ROW + EMBED + (size_t)head * 128 + c4 * 8);
    }
    // V tile needs a transpose -> manual load + LDS scatter
#pragma unroll
    for (int j = 0; j < 2; j++) {
      int q = tid + 256 * j;
      int key = q >> 4, c4 = q & 15;
      size_t vpos = segBase + (size_t)(kb + key) * dil;
      uint4 d = *(const uint4*)(qkv + vpos * ROW + 2 * EMBED + (size_t)head * 128 + c4 * 8);
      const u16* e = (const u16*)&d;
#pragma unroll
      for (int t = 0; t < 8; t++) Vt[(c4 * 8 + t) * PAD32 + key] = e[t];
    }
    async_wait0();
    __syncthreads();

    // scores: two 16x16 tiles over K=128 (4 chained WMMAs each)
    v8f s0 = v8f{}, s1 = v8f{};
#pragma unroll
    for (int kk = 0; kk < 4; kk++) {
      v16bf k0 = load_frag(Kt, PAD128, 0, kk * 32, lane);
      v16bf k1 = load_frag(Kt, PAD128, 16, kk * 32, lane);
      s0 = __builtin_amdgcn_wmma_f32_16x16x32_bf16(false, qf[kk], false, k0, (short)0, s0, false, false);
      s1 = __builtin_amdgcn_wmma_f32_16x16x32_bf16(false, qf[kk], false, k1, (short)0, s1, false, false);
    }

    // online softmax; rows live at (vgpr i, lane-half); reduce across the 16 N-lanes
#pragma unroll
    for (int i = 0; i < 8; i++) {
      float a = s0[i] * scale, b = s1[i] * scale;
      float mx = fmaxf(a, b);
#pragma unroll
      for (int m = 1; m < 16; m <<= 1) mx = fmaxf(mx, __shfl_xor(mx, m, 32));
      float mnew = fmaxf(rm[i], mx);
      float p0 = __expf(a - mnew), p1 = __expf(b - mnew);
      float ls = p0 + p1;
#pragma unroll
      for (int m = 1; m < 16; m <<= 1) ls += __shfl_xor(ls, m, 32);
      float corr = __expf(rm[i] - mnew);
      rl[i] = rl[i] * corr + ls;
      rm[i] = mnew;
#pragma unroll
      for (int n = 0; n < 8; n++) o[n][i] *= corr;
      u16* pr = &Pb[wave * 16 * PAD32 + (i + 8 * half) * PAD32 + l16];
      pr[0]  = f2bf(p0);
      pr[16] = f2bf(p1);
    }

    // P(16x32) @ V(32x128): A-frag from per-wave staging, B-frags from transposed V
    v16bf pf = load_frag(&Pb[wave * 16 * PAD32], PAD32, 0, 0, lane);
#pragma unroll
    for (int n = 0; n < 8; n++) {
      v16bf vf = load_frag(Vt, PAD32, n * 16, 0, lane);
      o[n] = __builtin_amdgcn_wmma_f32_16x16x32_bf16(false, pf, false, vf, (short)0, o[n], false, false);
    }
  }

  // normalize and scatter back to dilated positions (bf16)
#pragma unroll
  for (int i = 0; i < 8; i++) {
    const float inv = 1.0f / rl[i];
    const size_t pos = segBase + (size_t)(qrow0 + i + 8 * half) * dil;
    u16* op = out + pos * EMBED + (size_t)head * 128 + l16;
#pragma unroll
    for (int n = 0; n < 8; n++) op[n * 16] = f2bf(o[n][i] * inv);
  }
}

// ---------------------------------------------------------------- fused residual + LayerNorm
__global__ __launch_bounds__(256)
void ln_kernel(const float* __restrict__ a, const float* __restrict__ b,
               const float* __restrict__ g, const float* __restrict__ bb,
               float* __restrict__ y, u16* __restrict__ yb) {
  const int row = blockIdx.x;
  const float* pa = a + (size_t)row * EMBED;
  const float* pb = b + (size_t)row * EMBED;
  __shared__ float red[256];
  float s = 0.f;
  for (int i = threadIdx.x; i < EMBED; i += 256) s += pa[i] + pb[i];
  red[threadIdx.x] = s; __syncthreads();
  for (int o2 = 128; o2 > 0; o2 >>= 1) {
    if (threadIdx.x < o2) red[threadIdx.x] += red[threadIdx.x + o2];
    __syncthreads();
  }
  const float mean = red[0] * (1.0f / EMBED);
  __syncthreads();
  float v = 0.f;
  for (int i = threadIdx.x; i < EMBED; i += 256) {
    float d = pa[i] + pb[i] - mean; v += d * d;
  }
  red[threadIdx.x] = v; __syncthreads();
  for (int o2 = 128; o2 > 0; o2 >>= 1) {
    if (threadIdx.x < o2) red[threadIdx.x] += red[threadIdx.x + o2];
    __syncthreads();
  }
  const float rstd = rsqrtf(red[0] * (1.0f / EMBED) + 1e-5f);
  for (int i = threadIdx.x; i < EMBED; i += 256) {
    float o = (pa[i] + pb[i] - mean) * rstd * g[i] + bb[i];
    y[(size_t)row * EMBED + i] = o;
    if (yb) yb[(size_t)row * EMBED + i] = f2bf(o);
  }
}

// ---------------------------------------------------------------- host orchestration

extern "C" void kernel_launch(void* const* d_in, const int* in_sizes, int n_in,
                              void* d_out, int out_size, void* d_ws, size_t ws_size,
                              hipStream_t stream) {
  const float* x      = (const float*)d_in[0];
  const float* ee     = (const float*)d_in[1];
  const float* w_qkv  = (const float*)d_in[2];
  const float* b_qkv  = (const float*)d_in[3];
  const float* w_out  = (const float*)d_in[4];
  const float* b_out  = (const float*)d_in[5];
  const float* ln1g   = (const float*)d_in[6];
  const float* ln1b   = (const float*)d_in[7];
  const float* w1     = (const float*)d_in[8];
  const float* b1     = (const float*)d_in[9];
  const float* w2     = (const float*)d_in[10];
  const float* b2     = (const float*)d_in[11];
  const float* ln2g   = (const float*)d_in[12];
  const float* ln2b   = (const float*)d_in[13];
  float* out = (float*)d_out;
  char* ws = (char*)d_ws;

  const size_t SE  = (size_t)SEQ * EMBED;           // 25.2M
  const size_t o_wqkv = 0;
  const size_t o_wout = o_wqkv + (size_t)QKV_D * EMBED * 2;
  const size_t o_w1   = o_wout + (size_t)EMBED * EMBED * 2;
  const size_t o_w2   = o_w1   + (size_t)FFN_D * EMBED * 2;
  const size_t o_xb   = o_w2   + (size_t)EMBED * FFN_D * 2;   // bf16 [S,E]; reused as y1b
  const size_t o_xr   = o_xb   + SE * 2;                       // f32 residual
  const size_t o_qkv  = o_xr   + SE * 4;                       // bf16 [S,3E]
  const size_t o_attn = o_qkv  + (size_t)SEQ * QKV_D * 2;      // bf16 [S,E]
  const size_t o_proj = o_attn + SE * 2;                       // f32 [S,E]; reused as f2
  const size_t o_y1   = o_proj + SE * 4;                       // f32 [S,E]
  const size_t o_h1   = o_y1   + SE * 4;                       // bf16 [S,F]

  u16*   wqkvb = (u16*)(ws + o_wqkv);
  u16*   woutb = (u16*)(ws + o_wout);
  u16*   w1b   = (u16*)(ws + o_w1);
  u16*   w2b   = (u16*)(ws + o_w2);
  u16*   xb    = (u16*)(ws + o_xb);
  float* xr    = (float*)(ws + o_xr);
  u16*   qkv   = (u16*)(ws + o_qkv);
  u16*   attn  = (u16*)(ws + o_attn);
  float* proj  = (float*)(ws + o_proj);
  float* y1    = (float*)(ws + o_y1);
  u16*   y1b   = xb;    // xb dead after QKV GEMM
  u16*   h1    = (u16*)(ws + o_h1);
  float* f2    = proj;  // proj dead after LN1

  // 1) weight conversion fp32 -> bf16
  cvt_f32_bf16<<<(int)(((size_t)QKV_D * EMBED) / 1024), 256, 0, stream>>>(w_qkv, wqkvb, (size_t)QKV_D * EMBED);
  cvt_f32_bf16<<<(int)(((size_t)EMBED * EMBED) / 1024), 256, 0, stream>>>(w_out, woutb, (size_t)EMBED * EMBED);
  cvt_f32_bf16<<<(int)(((size_t)FFN_D * EMBED) / 1024), 256, 0, stream>>>(w1, w1b, (size_t)FFN_D * EMBED);
  cvt_f32_bf16<<<(int)(((size_t)EMBED * FFN_D) / 1024), 256, 0, stream>>>(w2, w2b, (size_t)EMBED * FFN_D);

  // 2) x + expert_embed (f32 residual + bf16 operand)
  add_embed_kernel<<<(int)(SE / 1024), 256, 0, stream>>>(x, ee, xr, xb);

  // 3) QKV GEMM: [8192,9216] = xb @ w_qkv^T
  gemm_bf16<0><<<dim3(QKV_D / 128, SEQ / 128), 256, 0, stream>>>(xb, wqkvb, b_qkv, qkv, SEQ, QKV_D, EMBED);

  // 4) dilated attention (zero unselected positions first)
  zero_u16_kernel<<<(int)(SE / 2048), 256, 0, stream>>>(attn, SE);
  attn_wmma<<<dim3(8, 8, 8), 256, 0, stream>>>(qkv, attn, 1024, 1, 0, 0);
  attn_wmma<<<dim3(8, 8, 4), 256, 0, stream>>>(qkv, attn, 2048, 2, 1, 8);
  attn_wmma<<<dim3(8, 8, 2), 256, 0, stream>>>(qkv, attn, 4096, 4, 2, 16);

  // 5) output projection (f32)
  gemm_bf16<1><<<dim3(EMBED / 128, SEQ / 128), 256, 0, stream>>>(attn, woutb, b_out, proj, SEQ, EMBED, EMBED);

  // 6) LN1(xr + proj) -> y1 (f32) + y1b (bf16)
  ln_kernel<<<SEQ, 256, 0, stream>>>(xr, proj, ln1g, ln1b, y1, y1b);

  // 7) FFN1 with exact GELU (bf16 out)
  gemm_bf16<2><<<dim3(FFN_D / 128, SEQ / 128), 256, 0, stream>>>(y1b, w1b, b1, h1, SEQ, FFN_D, EMBED);

  // 8) FFN2 (f32 out)
  gemm_bf16<1><<<dim3(EMBED / 128, SEQ / 128), 256, 0, stream>>>(h1, w2b, b2, f2, SEQ, EMBED, FFN_D);

  // 9) LN2(y1 + f2) -> d_out
  ln_kernel<<<SEQ, 256, 0, stream>>>(y1, f2, ln2g, ln2b, out, nullptr);
}
